// CausalSelfAttention_63608465653945
// MI455X (gfx1250) — matrix operand order repaired and checked by reading, not compiled, not run on hardware
//
#include <hip/hip_runtime.h>

// ---------------------------------------------------------------------------
// Types for CDNA5 WMMA (wave32): 16x16x32 bf16 -> f32
// ---------------------------------------------------------------------------
typedef __bf16 bf16_t;
typedef __attribute__((ext_vector_type(16))) __bf16 v16bf;
typedef __attribute__((ext_vector_type(8)))  __bf16 v8bf;
typedef __attribute__((ext_vector_type(4)))  __bf16 v4bf;
typedef __attribute__((ext_vector_type(8)))  float   v8f;
typedef __attribute__((ext_vector_type(4)))  int     v4i;

#define DIM      2048
#define SEQ      2048
#define BATCH    2
#define NHEADS   16
#define NKV      4
#define HD       128
#define NTOK     (BATCH * SEQ)     // 4096
#define KVDIM    (NKV * HD)        // 512

__device__ __forceinline__ bf16_t f2bf(float f) {
  union { float f; unsigned u; } a; a.f = f;
  unsigned r = a.u + 0x7FFFu + ((a.u >> 16) & 1u);  // round-nearest-even
  unsigned short h = (unsigned short)(r >> 16);
  bf16_t b; __builtin_memcpy(&b, &h, 2);
  return b;
}

__device__ __forceinline__ v8f wmma_bf16(v16bf a, v16bf b, v8f c) {
  return __builtin_amdgcn_wmma_f32_16x16x32_bf16(
      /*neg_a=*/false, a, /*neg_b=*/false, b,
      /*c_mod=*/(short)0, c, /*reuse_a=*/false, /*reuse_b=*/false);
}

// ---------------------------------------------------------------------------
// Async global->LDS copy (gfx1250). Builtin signature (from compiler
// diagnostic): (int4 AS1*, int4 AS3*, imm offset, imm cpol).
// ---------------------------------------------------------------------------
__device__ __forceinline__ void async_copy_b128(const bf16_t* gsrc, bf16_t* lds_dst) {
#if __has_builtin(__builtin_amdgcn_global_load_async_to_lds_b128)
  __builtin_amdgcn_global_load_async_to_lds_b128(
      (__attribute__((address_space(1))) v4i*)(const void*)gsrc,
      (__attribute__((address_space(3))) v4i*)(void*)lds_dst, 0, 0);
#else
  // Low 32 bits of a generic shared pointer are the LDS byte address
  unsigned lds_addr = (unsigned)(unsigned long long)lds_dst;
  asm volatile("global_load_async_to_lds_b128 %0, %1, off"
               :: "v"(lds_addr), "v"(gsrc) : "memory");
#endif
}

__device__ __forceinline__ void wait_async_all() {
#if __has_builtin(__builtin_amdgcn_s_wait_asynccnt)
  __builtin_amdgcn_s_wait_asynccnt(0);
#else
  asm volatile("s_wait_asynccnt 0x0" ::: "memory");
#endif
}

// ---------------------------------------------------------------------------
// Fragment loaders.
// A-fragment (16x32 bf16, M x K): lanes 0-15 hold row m, K {0..7,16..23};
// lanes 16-31 hold row m, K {8..15,24..31}.
// B-fragment (32x16 bf16, K x N): lane (n + 16*hl) holds column n with
// K {8*hl..8*hl+7, 16+8*hl..16+8*hl+7} (same order as A).
// ---------------------------------------------------------------------------
__device__ __forceinline__ v16bf load_a_frag(const bf16_t* base, int stride,
                                             int m0, int kk, int lane) {
  int hl = lane >> 4;
  int m  = m0 + (lane & 15);
  const bf16_t* p = base + (long)m * stride + kk + 8 * hl;
  v8bf lo = *(const v8bf*)p;
  v8bf hi = *(const v8bf*)(p + 16);
  v16bf r;
#pragma unroll
  for (int i = 0; i < 8; ++i) { r[i] = lo[i]; r[i + 8] = hi[i]; }
  return r;
}

__device__ __forceinline__ v16bf load_b_frag(const bf16_t* base, int stride,
                                             int n0, int kk, int lane) {
  int hl = lane >> 4;
  int n  = n0 + (lane & 15);
  const bf16_t* p = base + (long)n * stride + kk + 8 * hl;
  v8bf lo = *(const v8bf*)p;
  v8bf hi = *(const v8bf*)(p + 16);
  v16bf r;
#pragma unroll
  for (int i = 0; i < 8; ++i) { r[i] = lo[i]; r[i + 8] = hi[i]; }
  return r;
}

// ---------------------------------------------------------------------------
// 1) Blockwise fake-quant of weights: one wave per 128-elem block
// ---------------------------------------------------------------------------
__global__ void quant_weights(const float* __restrict__ W,
                              bf16_t* __restrict__ out, int nblocks) {
  int lane = threadIdx.x & 31;
  int blk  = blockIdx.x * (blockDim.x >> 5) + (threadIdx.x >> 5);
  if (blk >= nblocks) return;
  const float4 w = *(const float4*)(W + (long)blk * 128 + lane * 4);
  float amax = fmaxf(fmaxf(fabsf(w.x), fabsf(w.y)), fmaxf(fabsf(w.z), fabsf(w.w)));
#pragma unroll
  for (int m = 1; m < 32; m <<= 1) amax = fmaxf(amax, __shfl_xor(amax, m, 32));
  float s0 = fmaxf(amax * (1.0f / 31.0f), 1e-12f);
  _Float16 h = (_Float16)s0;                       // fp16 round-trip of scale
  float s = fmaxf((float)h, 6.103515625e-05f);     // clamp to fp16 tiny
  float inv = 1.0f / s;
  float vals[4] = {w.x, w.y, w.z, w.w};
  v4bf o;
#pragma unroll
  for (int j = 0; j < 4; ++j) {
    float q = rintf(fminf(fmaxf(vals[j] * inv, -32.0f), 31.0f));
    o[j] = f2bf(q * s);
  }
  *(v4bf*)(out + (long)blk * 128 + lane * 4) = o;
}

// ---------------------------------------------------------------------------
// 2) f32 -> bf16 conversion (x)
// ---------------------------------------------------------------------------
__global__ void cvt_f32_bf16(const float* __restrict__ in,
                             bf16_t* __restrict__ out, int n4) {
  int i = blockIdx.x * blockDim.x + threadIdx.x;
  if (i >= n4) return;
  float4 v = ((const float4*)in)[i];
  v4bf o; o[0] = f2bf(v.x); o[1] = f2bf(v.y); o[2] = f2bf(v.z); o[3] = f2bf(v.w);
  ((v4bf*)out)[i] = o;
}

// ---------------------------------------------------------------------------
// 3) Cooperative GEMM: C[m,n] = sum_k A[m,k] * B[n,k]  (bf16 in, f32 out)
//    Block = 8 waves = 128(M) x 64(N) macro tile. Per 32-wide k-step the
//    A(128x32) and B(64x32) tiles are staged in LDS via async-to-LDS DMA,
//    double-buffered so the next tile's DMA overlaps this tile's 32 WMMAs.
// ---------------------------------------------------------------------------
#define GEMM_BM 128
#define GEMM_BN 64
#define GEMM_BK 32

__global__ __launch_bounds__(256) void gemm_bf16(
    const bf16_t* __restrict__ A, const bf16_t* __restrict__ B,
    float* __restrict__ C, int M, int N, int K) {
  __shared__ bf16_t As[2][GEMM_BM * GEMM_BK];   // 2 x 8 KB
  __shared__ bf16_t Bs[2][GEMM_BN * GEMM_BK];   // 2 x 4 KB
  int tid  = threadIdx.x;
  int lane = tid & 31;
  int warp = tid >> 5;
  int nblk = N / GEMM_BN;
  int m0 = (blockIdx.x / nblk) * GEMM_BM;
  int n0 = (blockIdx.x % nblk) * GEMM_BN;

  // Stage one k-step: A is 512 x 16B chunks, B is 256 x 16B chunks.
  auto stage = [&](int kk, int buf) {
#pragma unroll
    for (int c = 0; c < 2; ++c) {
      int chunk = tid + c * 256;                 // 0..511
      int row = chunk >> 2;
      int cc  = (chunk & 3) * 8;                 // element offset within row
      async_copy_b128(A + (long)(m0 + row) * K + kk + cc,
                      &As[buf][row * GEMM_BK + cc]);
    }
    {
      int row = tid >> 2;
      int cc  = (tid & 3) * 8;
      async_copy_b128(B + (long)(n0 + row) * K + kk + cc,
                      &Bs[buf][row * GEMM_BK + cc]);
    }
  };

  stage(0, 0);
  wait_async_all();
  __syncthreads();

  v8f acc[4] = {};
  int mw = warp * 16;                            // wave's rows inside macro tile
  int buf = 0;
  for (int kk = 0; kk < K; kk += GEMM_BK, buf ^= 1) {
    if (kk + GEMM_BK < K) stage(kk + GEMM_BK, buf ^ 1);
    v16bf a = load_a_frag(&As[buf][0], GEMM_BK, mw, 0, lane);
#pragma unroll
    for (int f = 0; f < 4; ++f) {
      v16bf b = load_b_frag(&Bs[buf][0], GEMM_BK, f * 16, 0, lane);
      acc[f] = wmma_bf16(a, b, acc[f]);
    }
    wait_async_all();                            // own DMAs done before barrier
    __syncthreads();
  }

  int hl = lane >> 4, col = lane & 15;
#pragma unroll
  for (int f = 0; f < 4; ++f) {
    int n = n0 + f * 16 + col;
#pragma unroll
    for (int i = 0; i < 8; ++i) {
      int m = m0 + mw + i + 8 * hl;              // C-fragment row mapping
      C[(long)m * N + n] = acc[f][i];
    }
  }
}

// ---------------------------------------------------------------------------
// 4) Post-QKV: RMSNorm(head) + RoPE + q_gain; writes bf16 Q/K (d-contig) and
//    V transposed (d-major). One wave per (token, unit).
// ---------------------------------------------------------------------------
__global__ void qkv_post(const float* __restrict__ qf,
                         const float* __restrict__ kf,
                         const float* __restrict__ vf,
                         const float* __restrict__ q_gain,
                         bf16_t* __restrict__ Qb,
                         bf16_t* __restrict__ Kb,
                         bf16_t* __restrict__ Vt) {
  int lane = threadIdx.x & 31;
  int wave = blockIdx.x * (blockDim.x >> 5) + (threadIdx.x >> 5);
  int unit = wave % 24;
  int tok  = wave / 24;
  if (tok >= NTOK) return;
  int b = tok / SEQ, spos = tok % SEQ;
  int d0 = lane * 4;
  float vals[4];

  if (unit < 20) {  // q or k: rmsnorm + rope
    int h   = (unit < 16) ? unit : (unit - 16);
    const float* p = (unit < 16)
        ? (qf + (long)tok * DIM + h * HD + d0)
        : (kf + (long)tok * KVDIM + h * HD + d0);
    float4 w = *(const float4*)p;
    vals[0] = w.x; vals[1] = w.y; vals[2] = w.z; vals[3] = w.w;
    float ss = vals[0]*vals[0] + vals[1]*vals[1] + vals[2]*vals[2] + vals[3]*vals[3];
#pragma unroll
    for (int m = 1; m < 32; m <<= 1) ss += __shfl_xor(ss, m, 32);
    float r = __frsqrt_rn(ss * (1.0f / 128.0f) + 1.1920929e-07f);
#pragma unroll
    for (int j = 0; j < 4; ++j) vals[j] *= r;
    // RoPE: positions d and d^64 live in lanes L and L^16
    float other[4];
#pragma unroll
    for (int j = 0; j < 4; ++j) other[j] = __shfl_xor(vals[j], 16, 32);
#pragma unroll
    for (int j = 0; j < 4; ++j) {
      int d = d0 + j;
      int i = d & 63;
      float invf = __expf(-(float)(2 * i) * (9.210340371976184f / 128.0f));
      float ang = (float)spos * invf;
      float sn, cs; __sincosf(ang, &sn, &cs);
      if (d < 64) vals[j] = vals[j] * cs + other[j] * sn;
      else        vals[j] = -other[j] * sn + vals[j] * cs;
    }
    v4bf o;
    if (unit < 16) {
      float g = q_gain[h];
#pragma unroll
      for (int j = 0; j < 4; ++j) o[j] = f2bf(vals[j] * g);
      *(v4bf*)(Qb + (((long)(b * NHEADS + h) * SEQ + spos) * HD + d0)) = o;
    } else {
#pragma unroll
      for (int j = 0; j < 4; ++j) o[j] = f2bf(vals[j]);
      *(v4bf*)(Kb + (((long)(b * NKV + h) * SEQ + spos) * HD + d0)) = o;
    }
  } else {          // v: transpose-store (d-major)
    int kv = unit - 20;
    float4 w = *(const float4*)(vf + (long)tok * KVDIM + kv * HD + d0);
    vals[0] = w.x; vals[1] = w.y; vals[2] = w.z; vals[3] = w.w;
#pragma unroll
    for (int j = 0; j < 4; ++j) {
      int d = d0 + j;
      Vt[((long)(b * NKV + kv) * HD + d) * SEQ + spos] = f2bf(vals[j]);
    }
  }
}

// ---------------------------------------------------------------------------
// 5) Flash attention: one wave per (b, h, 16-row q-tile). Online softmax on
//    C-fragment layout; P re-laid out C->A through per-wave LDS tile.
// ---------------------------------------------------------------------------
__global__ void flash_attn(const bf16_t* __restrict__ Qb,
                           const bf16_t* __restrict__ Kb,
                           const bf16_t* __restrict__ Vt,
                           bf16_t* __restrict__ Ob) {
  __shared__ float stage[4][16 * 32];   // per-wave P staging (16 rows x 32 keys)
  int lane = threadIdx.x & 31;
  int warp = threadIdx.x >> 5;
  int wave = blockIdx.x * 4 + warp;
  int qt = wave & 127;
  int bh = wave >> 7;
  if (bh >= BATCH * NHEADS) return;
  int h = bh & 15, b = bh >> 4;
  int kvh = h >> 2;
  int q0 = qt * 16;
  int hl = lane >> 4, col = lane & 15;

  const bf16_t* Qp = Qb + (long)(b * NHEADS + h) * SEQ * HD;
  const bf16_t* Kp = Kb + (long)(b * NKV + kvh) * SEQ * HD;
  const bf16_t* Vp = Vt + (long)(b * NKV + kvh) * HD * SEQ;

  v16bf qfrag[4];
#pragma unroll
  for (int f = 0; f < 4; ++f) qfrag[f] = load_a_frag(Qp, HD, q0, f * 32, lane);

  v8f o[8] = {};
  float mi[8], li[8];
#pragma unroll
  for (int i = 0; i < 8; ++i) { mi[i] = -3.0e38f; li[i] = 0.0f; }
  const float scale = 0.08838834764831845f;  // 1/sqrt(128)

  for (int kb = 0; kb < q0 + 16; kb += 32) {
    // ---- S = Q K^T over 32 keys (two 16x16 fragments) ----
    v8f s0 = {}, s1 = {};
#pragma unroll
    for (int kk = 0; kk < 4; ++kk) {
      v16bf b0 = load_b_frag(Kp, HD, kb,      kk * 32, lane);
      v16bf b1 = load_b_frag(Kp, HD, kb + 16, kk * 32, lane);
      s0 = wmma_bf16(qfrag[kk], b0, s0);
      s1 = wmma_bf16(qfrag[kk], b1, s1);
    }
    // ---- causal mask + scale + row max ----
    float mnew[8];
#pragma unroll
    for (int i = 0; i < 8; ++i) {
      int row = q0 + i + 8 * hl;
      int k0 = kb + col, k1 = kb + 16 + col;
      float a = (k0 <= row) ? s0[i] * scale : -3.0e38f;
      float c = (k1 <= row) ? s1[i] * scale : -3.0e38f;
      s0[i] = a; s1[i] = c;
      float mx = fmaxf(a, c);
#pragma unroll
      for (int m = 1; m < 16; m <<= 1) mx = fmaxf(mx, __shfl_xor(mx, m, 32));
      mnew[i] = fmaxf(mi[i], mx);
    }
    // ---- exp, row sums, rescale accumulators ----
    float corr[8];
#pragma unroll
    for (int i = 0; i < 8; ++i) {
      float p0 = __expf(s0[i] - mnew[i]);
      float p1 = __expf(s1[i] - mnew[i]);
      s0[i] = p0; s1[i] = p1;
      float sum = p0 + p1;
#pragma unroll
      for (int m = 1; m < 16; m <<= 1) sum += __shfl_xor(sum, m, 32);
      corr[i] = __expf(mi[i] - mnew[i]);
      li[i] = li[i] * corr[i] + sum;
      mi[i] = mnew[i];
    }
#pragma unroll
    for (int f = 0; f < 8; ++f)
#pragma unroll
      for (int i = 0; i < 8; ++i) o[f][i] *= corr[i];
    // ---- stage P (C layout) -> LDS -> reload as A fragment ----
    float* st = stage[warp];
#pragma unroll
    for (int i = 0; i < 8; ++i) {
      int row = i + 8 * hl;
      st[row * 32 + col]      = s0[i];
      st[row * 32 + 16 + col] = s1[i];
    }
    v16bf pfrag;
    {
      int row = lane & 15;
      int c0 = 8 * hl, c1 = 16 + 8 * hl;
#pragma unroll
      for (int j = 0; j < 8; ++j) {
        pfrag[j]     = f2bf(st[row * 32 + c0 + j]);
        pfrag[j + 8] = f2bf(st[row * 32 + c1 + j]);
      }
    }
    // ---- O += P * V (V is d-major, so B fragments are contiguous) ----
#pragma unroll
    for (int f = 0; f < 8; ++f) {
      v16bf bv = load_b_frag(Vp, SEQ, f * 16, kb, lane);
      o[f] = wmma_bf16(pfrag, bv, o[f]);
    }
  }
  // ---- normalize and store O as bf16 [token][h*128+d] ----
#pragma unroll
  for (int i = 0; i < 8; ++i) li[i] = 1.0f / li[i];
#pragma unroll
  for (int f = 0; f < 8; ++f)
#pragma unroll
    for (int i = 0; i < 8; ++i) {
      int tok = b * SEQ + q0 + i + 8 * hl;
      int d   = h * HD + f * 16 + col;
      Ob[(long)tok * DIM + d] = f2bf(o[f][i] * li[i]);
    }
}

// ---------------------------------------------------------------------------
// Launch
// ---------------------------------------------------------------------------
extern "C" void kernel_launch(void* const* d_in, const int* in_sizes, int n_in,
                              void* d_out, int out_size, void* d_ws, size_t ws_size,
                              hipStream_t stream) {
  const float* x     = (const float*)d_in[0];
  const float* Wq    = (const float*)d_in[1];
  const float* Wk    = (const float*)d_in[2];
  const float* Wv    = (const float*)d_in[3];
  const float* Wp    = (const float*)d_in[4];
  const float* qgain = (const float*)d_in[5];
  float* out = (float*)d_out;

  char* ws = (char*)d_ws;
  size_t off = 0;
  bf16_t* x_bf  = (bf16_t*)(ws + off); off += (size_t)NTOK * DIM * 2;      // 16 MB
  bf16_t* wq_bf = (bf16_t*)(ws + off); off += (size_t)DIM * DIM * 2;       //  8 MB
  bf16_t* wk_bf = (bf16_t*)(ws + off); off += (size_t)KVDIM * DIM * 2;     //  2 MB
  bf16_t* wv_bf = (bf16_t*)(ws + off); off += (size_t)KVDIM * DIM * 2;     //  2 MB
  bf16_t* wp_bf = (bf16_t*)(ws + off); off += (size_t)DIM * DIM * 2;       //  8 MB
  float*  qf    = (float*)(ws + off);  off += (size_t)NTOK * DIM * 4;      // 32 MB
  float*  kf    = (float*)(ws + off);  off += (size_t)NTOK * KVDIM * 4;    //  8 MB
  float*  vfb   = (float*)(ws + off);  off += (size_t)NTOK * KVDIM * 4;    //  8 MB
  bf16_t* Qb    = (bf16_t*)(ws + off); off += (size_t)NTOK * DIM * 2;      // 16 MB
  bf16_t* Kb    = (bf16_t*)(ws + off); off += (size_t)NTOK * KVDIM * 2;    //  4 MB
  bf16_t* Vt    = (bf16_t*)(ws + off); off += (size_t)NTOK * KVDIM * 2;    //  4 MB
  bf16_t* Ob    = (bf16_t*)(ws + off); off += (size_t)NTOK * DIM * 2;      // 16 MB

  // 1) quantize weights -> bf16
  {
    int nb;
    nb = DIM * (DIM / 128);
    quant_weights<<<(nb + 3) / 4, 128, 0, stream>>>(Wq, wq_bf, nb);
    nb = KVDIM * (DIM / 128);
    quant_weights<<<(nb + 3) / 4, 128, 0, stream>>>(Wk, wk_bf, nb);
    quant_weights<<<(nb + 3) / 4, 128, 0, stream>>>(Wv, wv_bf, nb);
    nb = DIM * (DIM / 128);
    quant_weights<<<(nb + 3) / 4, 128, 0, stream>>>(Wp, wp_bf, nb);
  }
  // 2) x -> bf16
  {
    int n4 = NTOK * DIM / 4;
    cvt_f32_bf16<<<(n4 + 255) / 256, 256, 0, stream>>>(x, x_bf, n4);
  }
  // 3) QKV GEMMs (f32 accum out), async-LDS double-buffered
  {
    int blocks = (NTOK / GEMM_BM) * (DIM / GEMM_BN);     // 32*32 = 1024
    gemm_bf16<<<blocks, 256, 0, stream>>>(x_bf, wq_bf, qf, NTOK, DIM, DIM);
    blocks = (NTOK / GEMM_BM) * (KVDIM / GEMM_BN);       // 32*8 = 256
    gemm_bf16<<<blocks, 256, 0, stream>>>(x_bf, wk_bf, kf, NTOK, KVDIM, DIM);
    gemm_bf16<<<blocks, 256, 0, stream>>>(x_bf, wv_bf, vfb, NTOK, KVDIM, DIM);
  }
  // 4) RMSNorm + RoPE + gain; layout for attention
  {
    int waves = NTOK * 24;                               // 98304
    qkv_post<<<waves / 4, 128, 0, stream>>>(qf, kf, vfb, qgain, Qb, Kb, Vt);
  }
  // 5) causal flash attention
  {
    int waves = BATCH * NHEADS * (SEQ / 16);             // 4096
    flash_attn<<<waves / 4, 128, 0, stream>>>(Qb, Kb, Vt, Ob);
  }
  // 6) output projection -> f32 out
  {
    int blocks = (NTOK / GEMM_BM) * (DIM / GEMM_BN);     // 1024
    gemm_bf16<<<blocks, 256, 0, stream>>>(Ob, wp_bf, out, NTOK, DIM, DIM);
  }
  (void)in_sizes; (void)n_in; (void)out_size; (void)ws_size;
}
